// BidirRecurrentModel_88708254531704
// MI455X (gfx1250) — matrix operand order repaired and checked by reading, not compiled, not run on hardware
//
#include <hip/hip_runtime.h>
#include <hip/hip_bf16.h>
#include <stdint.h>

// Problem constants (from the reference)
#define T_LEN 256
#define B_SZ  64
#define H_SZ  500
#define E_SZ  6
#define KP    512            // K (=H) padded to a multiple of 32 for bf16 WMMA
#define NO    1500           // 3*H gate width
#define NOP   1504           // padded to multiple of 16
#define NT_N  (NOP / 16)     // 94 N-tiles
#define EXPERT_DIV 10000
#define NBLK  4              // cooperating blocks per direction in the scan
#define WPB   8              // waves per 256-thread block (wave32)
#define MAXMT 10             // max padded expert M-tiles for B=64, E=6

typedef __attribute__((ext_vector_type(16))) __bf16         v16bf;
typedef __attribute__((ext_vector_type(8)))  float          v8f;
typedef __attribute__((ext_vector_type(8)))  unsigned short v8us;

union Frag {            // 32 bytes: one WMMA 16-bit A or B fragment
  v8us  h[2];
  v16bf v;
};

__device__ __forceinline__ unsigned short f2bf(float f) {   // RNE fp32->bf16
  unsigned u = __float_as_uint(f);
  u += 0x7fffu + ((u >> 16) & 1u);
  return (unsigned short)(u >> 16);
}

__device__ __forceinline__ float sigmoidf_(float x) {
  return 1.0f / (1.0f + __expf(-x));
}

// ---------------------------------------------------------------------------
// Per-timestep expert grouping: sort batch rows by expert id, pad groups to 16.
// B=64 -> at most 10 M-tiles.  Done redundantly by thread 0 of each block.
// ---------------------------------------------------------------------------
__device__ __forceinline__ void build_groups(const int* tokens, int t_eff,
                                             int* s_map, int* s_texp, int* s_ntiles) {
  if (threadIdx.x == 0) {
    int cnt[E_SZ];
    int lists[E_SZ][B_SZ];
    for (int e = 0; e < E_SZ; ++e) cnt[e] = 0;
    for (int b = 0; b < B_SZ; ++b) {
      int tok = tokens[t_eff * B_SZ + b];
      int e = tok / EXPERT_DIV;
      e = (e < 0) ? 0 : ((e >= E_SZ) ? (E_SZ - 1) : e);
      lists[e][cnt[e]++] = b;
    }
    int nt = 0;
    for (int e = 0; e < E_SZ; ++e) {
      for (int base = 0; base < cnt[e]; base += 16) {
        s_texp[nt] = e;
        for (int r = 0; r < 16; ++r)
          s_map[nt * 16 + r] = (base + r < cnt[e]) ? lists[e][base + r] : -1;
        ++nt;
      }
    }
    *s_ntiles = nt;
  }
  __syncthreads();
}

// ---------------------------------------------------------------------------
// One wave sweeps its share of the 94 N-tiles, TWO tiles at a time, sharing
// one A fragment between two independent accumulator chains.  A (16x32 bf16)
// is hoisted into VGPRs; the two B streams (32x16 bf16 each) come from the
// L2-resident bf16 weights as paired global_load_b128 clauses whose latency
// overlaps the partner chain's v_wmma.
// ---------------------------------------------------------------------------
__device__ __forceinline__ void wmma_ntiles(const unsigned short* sA,     // [16][KP] staged A rows
                                            const unsigned short* Wbe,    // expert weights [NOP][KP]
                                            const float* bias_e,          // [NO]
                                            float* outBase,               // [B_SZ][NOP] fp32 gates
                                            const int* mapRow,            // 16 batch indices (-1 = pad)
                                            int ntStart, int ntStride) {
  const int lane = threadIdx.x & 31;
  const int half = lane >> 4;      // lane group 0/1
  const int nn   = lane & 15;
  const int aBase = nn * KP + half * 16;  // A: row = lane%16, K chunk per half

  for (int nt = ntStart; nt < NT_N; nt += 2 * ntStride) {
    const int  ntB  = nt + ntStride;
    const bool hasB = ntB < NT_N;
    const int  colA = nt * 16 + nn;                      // output column (= weight row)
    const int  colB = (hasB ? ntB : nt) * 16 + nn;       // alias A when no 2nd tile

    // Unconditional clamped bias reads (pad columns are never consumed).
    const float bvA = bias_e[(colA < NO) ? colA : (NO - 1)];
    const float bvB = bias_e[(colB < NO) ? colB : (NO - 1)];
    v8f accA, accB;
#pragma unroll
    for (int i = 0; i < 8; ++i) {
      accA[i] = (colA < NO) ? bvA : 0.0f;
      accB[i] = (colB < NO) ? bvB : 0.0f;
    }

    const unsigned short* BpA = Wbe + (size_t)colA * KP + half * 16;
    const unsigned short* BpB = Wbe + (size_t)colB * KP + half * 16;

#pragma unroll
    for (int k0 = 0; k0 < KP; k0 += 32) {
      Frag a, b0, b1;
      b0.h[0] = *(const v8us*)(BpA + k0);                // K = k0+half*16 .. +7
      b0.h[1] = *(const v8us*)(BpA + k0 + 8);            // K = .. +8..15
      b1.h[0] = *(const v8us*)(BpB + k0);
      b1.h[1] = *(const v8us*)(BpB + k0 + 8);
      a.h[0] = *(const v8us*)(sA + aBase + k0);          // K = k0+half*16 .. +7
      a.h[1] = *(const v8us*)(sA + aBase + k0 + 8);      // K = .. +8..15
      accA = __builtin_amdgcn_wmma_f32_16x16x32_bf16(
                 false, a.v, false, b0.v, (short)0, accA, false, false);
      accB = __builtin_amdgcn_wmma_f32_16x16x32_bf16(
                 false, a.v, false, b1.v, (short)0, accB, false, false);
    }
#pragma unroll
    for (int r = 0; r < 8; ++r) {                        // D layout: M = r + 8*half
      const int m = r + half * 8;
      const int bidx = mapRow[m];
      if (bidx >= 0) {
        outBase[(size_t)bidx * NOP + colA] = accA[r];
        if (hasB) outBase[(size_t)bidx * NOP + colB] = accB[r];
      }
    }
  }
}

// Stage 16 gathered rows of a [B_SZ][KP] bf16 state buffer into LDS.
__device__ __forceinline__ void stage_rows_bf(const unsigned short* srcBf,
                                              const int* mapRow, unsigned short* sA) {
  for (int c = threadIdx.x; c < 16 * (KP / 8); c += 256) {
    const int r = c >> 6;                 // KP/8 == 64 chunks per row
    const int off = (c & 63) << 3;
    const int b = mapRow[r];
    v8us val = {0, 0, 0, 0, 0, 0, 0, 0};
    if (b >= 0) val = *(const v8us*)(srcBf + (size_t)b * KP + off);
    *(v8us*)(sA + r * KP + off) = val;
  }
}

// Device-scope sense-reversing barrier across the NBLK blocks of one direction.
__device__ __forceinline__ void dir_barrier(unsigned* cnt, volatile unsigned* gen,
                                            unsigned nblk, unsigned* s_genp) {
  __syncthreads();
  __threadfence();
  if (threadIdx.x == 0) {
    const unsigned target = ++(*s_genp);
    const unsigned prev = atomicAdd(cnt, 1u);
    if (prev == nblk - 1u) {
      atomicExch(cnt, 0u);
      atomicAdd((unsigned*)gen, 1u);
    } else {
      while (*gen < target) __builtin_amdgcn_s_sleep(2);
    }
  }
  __syncthreads();
  __threadfence();
}

// ---------------------------------------------------------------------------
// Kernel 1a: zero h-states (fp32 + padded bf16) and barrier counters.
// ---------------------------------------------------------------------------
__global__ void __launch_bounds__(256) k_init(float* stateF, size_t nf,
                                              unsigned short* stateU, size_t nu,
                                              unsigned* bars) {
  const size_t stride = (size_t)gridDim.x * blockDim.x;
  for (size_t i = (size_t)blockIdx.x * blockDim.x + threadIdx.x; i < nf; i += stride)
    stateF[i] = 0.0f;
  for (size_t i = (size_t)blockIdx.x * blockDim.x + threadIdx.x; i < nu; i += stride)
    stateU[i] = 0;
  if (blockIdx.x == 0 && threadIdx.x < 4) bars[threadIdx.x] = 0u;
}

// ---------------------------------------------------------------------------
// Kernel 1b: convert the 4 weight tensors fp32 [E,1500,500] -> bf16 [E,1504,512]
// (zero padded), contiguous: [ih0 | hh0 | ih1 | hh1].  ~37MB -> lives in L2.
// ---------------------------------------------------------------------------
__global__ void __launch_bounds__(256) k_convert(const float* W_ih0, const float* W_hh0,
                                                 const float* W_ihL, const float* W_hhL,
                                                 unsigned short* Wb) {
  const size_t CH = (size_t)E_SZ * NOP * KP;
  const size_t idx = (size_t)blockIdx.x * blockDim.x + threadIdx.x;
  if (idx >= 4 * CH) return;
  const int tsr = (int)(idx / CH);
  const size_t r0 = idx % CH;
  const int e   = (int)(r0 / ((size_t)NOP * KP));
  const int row = (int)((r0 / KP) % NOP);
  const int col = (int)(r0 & (KP - 1));
  const float* src = (tsr == 0) ? W_ih0 : (tsr == 1) ? W_hh0 : (tsr == 2) ? W_ihL : W_hhL;
  float v = 0.0f;
  if (row < NO && col < H_SZ)
    v = src[((size_t)e * NO + row) * H_SZ + col];
  Wb[idx] = f2bf(v);
}

// ---------------------------------------------------------------------------
// Kernel 2: precompute layer-0 input gates for ALL timesteps (parallel over t):
//   gi0[t,b,:] = bf16(emb[tok[t,b]]) @ W_ih0[e_b]^T + b_ih0[e_b]
// One block per t; expert-grouped WMMA.
// ---------------------------------------------------------------------------
__global__ void __launch_bounds__(256) k_gi0(const int* tokens, const float* emb,
                                             const unsigned short* Wb_ih0,
                                             const float* b_ih0, float* gi0) {
  __shared__ __align__(16) unsigned short sA[16 * KP];
  __shared__ int s_map[MAXMT * 16];
  __shared__ int s_texp[MAXMT];
  __shared__ int s_ntiles;

  const int t = blockIdx.x;
  build_groups(tokens, t, s_map, s_texp, &s_ntiles);
  float* outT = gi0 + (size_t)t * B_SZ * NOP;
  const int wave = threadIdx.x >> 5;

  for (int mt = 0; mt < s_ntiles; ++mt) {
    // Stage 16 embedding rows (gather by token), converting fp32 -> bf16, zero-pad K.
    for (int idx = threadIdx.x; idx < 16 * KP; idx += 256) {
      const int r = idx >> 9;
      const int c = idx & (KP - 1);
      const int b = s_map[mt * 16 + r];
      float v = 0.0f;
      if (b >= 0 && c < H_SZ) {
        const int tok = tokens[(size_t)t * B_SZ + b];
        v = emb[(size_t)tok * H_SZ + c];
      }
      sA[idx] = f2bf(v);
    }
    __syncthreads();
    const int e = s_texp[mt];
    wmma_ntiles(sA, Wb_ih0 + (size_t)e * NOP * KP, b_ih0 + (size_t)e * NO,
                outT, s_map + mt * 16, wave, WPB);
    __syncthreads();
  }
}

// ---------------------------------------------------------------------------
// Kernel 3: persistent bidirectional scan.  grid = 2 directions * NBLK blocks.
// Per step: gh0 GEMM | barrier | layer0 elementwise | barrier |
//           gi1+gh1 GEMMs | barrier | layer1 elementwise + output | barrier.
// ---------------------------------------------------------------------------
__device__ __forceinline__ void matmul_phase(const unsigned short* srcBf,
                                             const unsigned short* Wb, const float* bias,
                                             float* outBuf, int ntiles,
                                             const int* s_map, const int* s_texp,
                                             unsigned short* sA, int blk) {
  const int wave = threadIdx.x >> 5;
  for (int mt = 0; mt < ntiles; ++mt) {
    stage_rows_bf(srcBf, s_map + mt * 16, sA);
    __syncthreads();
    const int e = s_texp[mt];
    wmma_ntiles(sA, Wb + (size_t)e * NOP * KP, bias + (size_t)e * NO,
                outBuf, s_map + mt * 16, blk * WPB + wave, NBLK * WPB);
    __syncthreads();
  }
}

__global__ void __launch_bounds__(256) k_scan(const int* tokens, const float* gi0,
    const unsigned short* Wb_hh0, const unsigned short* Wb_ih1, const unsigned short* Wb_hh1,
    const float* b_hh0, const float* b_ihL, const float* b_hhL,
    float* h0f_all, float* h1f_all, unsigned short* h0b_all, unsigned short* h1b_all,
    float* gh0_all, float* gi1_all, float* gh1_all,
    float* d_out, float* outB, unsigned* bars) {
  __shared__ __align__(16) unsigned short sA[16 * KP];
  __shared__ int s_map[MAXMT * 16];
  __shared__ int s_texp[MAXMT];
  __shared__ int s_ntiles;
  __shared__ unsigned s_gen;

  const int tid = threadIdx.x;
  const int dir = blockIdx.x / NBLK;
  const int blk = blockIdx.x % NBLK;
  const size_t BH = (size_t)B_SZ * H_SZ;

  float*          h0f = h0f_all + (size_t)dir * BH;
  float*          h1f = h1f_all + (size_t)dir * BH;
  unsigned short* h0b = h0b_all + (size_t)dir * B_SZ * KP;
  unsigned short* h1b = h1b_all + (size_t)dir * B_SZ * KP;
  float*          gh0 = gh0_all + (size_t)dir * B_SZ * NOP;
  float*          gi1 = gi1_all + (size_t)dir * B_SZ * NOP;
  float*          gh1 = gh1_all + (size_t)dir * B_SZ * NOP;
  unsigned* cnt = bars + dir * 2;
  unsigned* gen = bars + dir * 2 + 1;

  if (tid == 0) s_gen = 0;
  __syncthreads();

  for (int t = 0; t < T_LEN; ++t) {
    const int t_eff = dir ? (T_LEN - 1 - t) : t;
    build_groups(tokens, t_eff, s_map, s_texp, &s_ntiles);

    // Phase A: gh0 = h0 @ W_hh0[e]^T + b_hh0[e]
    matmul_phase(h0b, Wb_hh0, b_hh0, gh0, s_ntiles, s_map, s_texp, sA, blk);
    dir_barrier(cnt, gen, NBLK, &s_gen);

    // Phase B: layer-0 GRU elementwise; gi0 was precomputed.
    for (int idx = blk * 256 + tid; idx < (int)BH; idx += NBLK * 256) {
      const int b = idx / H_SZ, o = idx % H_SZ;
      const int tok = tokens[t_eff * B_SZ + b];
      const float* gi = gi0 + ((size_t)t_eff * B_SZ + b) * NOP;
      const float* gh = gh0 + (size_t)b * NOP;
      const float r = sigmoidf_(gi[o] + gh[o]);
      const float z = sigmoidf_(gi[o + H_SZ] + gh[o + H_SZ]);
      const float n = tanhf(gi[o + 2 * H_SZ] + r * gh[o + 2 * H_SZ]);
      const float hp = h0f[idx];
      const float hv = (tok > 0) ? ((1.0f - z) * n + z * hp) : hp;
      h0f[idx] = hv;
      h0b[(size_t)b * KP + o] = f2bf(hv);
    }
    dir_barrier(cnt, gen, NBLK, &s_gen);

    // Phase C: gi1 = h0_new @ W_ihL^T + b_ihL ; gh1 = h1 @ W_hhL^T + b_hhL
    matmul_phase(h0b, Wb_ih1, b_ihL, gi1, s_ntiles, s_map, s_texp, sA, blk);
    matmul_phase(h1b, Wb_hh1, b_hhL, gh1, s_ntiles, s_map, s_texp, sA, blk);
    dir_barrier(cnt, gen, NBLK, &s_gen);

    // Phase D: layer-1 GRU elementwise + output contribution.
    for (int idx = blk * 256 + tid; idx < (int)BH; idx += NBLK * 256) {
      const int b = idx / H_SZ, o = idx % H_SZ;
      const int tok = tokens[t_eff * B_SZ + b];
      const float* gi = gi1 + (size_t)b * NOP;
      const float* gh = gh1 + (size_t)b * NOP;
      const float r = sigmoidf_(gi[o] + gh[o]);
      const float z = sigmoidf_(gi[o + H_SZ] + gh[o + H_SZ]);
      const float n = tanhf(gi[o + 2 * H_SZ] + r * gh[o + 2 * H_SZ]);
      const float hp = h1f[idx];
      const float hv = (tok > 0) ? ((1.0f - z) * n + z * hp) : hp;
      h1f[idx] = hv;
      h1b[(size_t)b * KP + o] = f2bf(hv);
      const float contrib = 0.5f * hv;
      if (dir == 0)  // out[s] = 0.5*(h_fwd[T-1-s] + h_bwd[s]); fwd step t -> s = T-1-t
        d_out[((size_t)(T_LEN - 1 - t) * B_SZ + b) * H_SZ + o] = contrib;
      else
        outB[((size_t)t * B_SZ + b) * H_SZ + o] = contrib;
    }
    dir_barrier(cnt, gen, NBLK, &s_gen);
  }
}

// ---------------------------------------------------------------------------
// Kernel 4: combine backward contribution + emit final hidden states.
// d_out layout: [T,B,H] output ++ [4,B,H] = [h_f_l0, h_f_l1, h_b_l0, h_b_l1].
// ---------------------------------------------------------------------------
__global__ void __launch_bounds__(256) k_combine(float* d_out, const float* outB,
                                                 const float* h0f_all, const float* h1f_all) {
  const size_t BH = (size_t)B_SZ * H_SZ;
  const size_t N = (size_t)T_LEN * BH;
  const size_t idx = (size_t)blockIdx.x * blockDim.x + threadIdx.x;
  if (idx < N) d_out[idx] += outB[idx];
  if (idx < BH) {
    d_out[N + idx]          = h0f_all[idx];        // forward layer 0
    d_out[N + BH + idx]     = h1f_all[idx];        // forward layer 1
    d_out[N + 2 * BH + idx] = h0f_all[BH + idx];   // backward layer 0
    d_out[N + 3 * BH + idx] = h1f_all[BH + idx];   // backward layer 1
  }
}

// ---------------------------------------------------------------------------
extern "C" void kernel_launch(void* const* d_in, const int* in_sizes, int n_in,
                              void* d_out, int out_size, void* d_ws, size_t ws_size,
                              hipStream_t stream) {
  (void)in_sizes; (void)n_in; (void)out_size; (void)ws_size;

  const int*   tokens = (const int*)  d_in[0];
  const float* emb    = (const float*)d_in[1];
  const float* W_ih0  = (const float*)d_in[2];
  const float* W_hh0  = (const float*)d_in[3];
  const float* b_ih0  = (const float*)d_in[4];
  const float* b_hh0  = (const float*)d_in[5];
  const float* W_ihL  = (const float*)d_in[6];
  const float* W_hhL  = (const float*)d_in[7];
  const float* b_ihL  = (const float*)d_in[8];
  const float* b_hhL  = (const float*)d_in[9];
  float* out = (float*)d_out;

  // Workspace layout
  const size_t CH = (size_t)E_SZ * NOP * KP;      // elems per bf16 weight tensor
  const size_t BH = (size_t)B_SZ * H_SZ;
  char* ws = (char*)d_ws;
  size_t off = 0;
  auto take = [&](size_t bytes) { size_t o = off; off += (bytes + 255) & ~(size_t)255; return o; };

  unsigned short* Wb     = (unsigned short*)(ws + take(4 * CH * sizeof(unsigned short)));
  float*          gi0    = (float*)(ws + take((size_t)T_LEN * B_SZ * NOP * sizeof(float)));
  float*          outB   = (float*)(ws + take((size_t)T_LEN * BH * sizeof(float)));
  float*          stateF = (float*)(ws + take(4 * BH * sizeof(float)));
  unsigned short* stateU = (unsigned short*)(ws + take((size_t)4 * B_SZ * KP * sizeof(unsigned short)));
  float*          gh0a   = (float*)(ws + take((size_t)2 * B_SZ * NOP * sizeof(float)));
  float*          gi1a   = (float*)(ws + take((size_t)2 * B_SZ * NOP * sizeof(float)));
  float*          gh1a   = (float*)(ws + take((size_t)2 * B_SZ * NOP * sizeof(float)));
  unsigned*       bars   = (unsigned*)(ws + take(64));

  float*          h0f_all = stateF;                 // [2][B,H]
  float*          h1f_all = stateF + 2 * BH;        // [2][B,H]
  unsigned short* h0b_all = stateU;                 // [2][B,KP] bf16
  unsigned short* h1b_all = stateU + 2 * (size_t)B_SZ * KP;

  // 1) init state + barriers; convert weights to padded bf16 (L2-resident).
  k_init<<<512, 256, 0, stream>>>(stateF, 4 * BH, stateU, (size_t)4 * B_SZ * KP, bars);
  const size_t convN = 4 * CH;
  k_convert<<<(unsigned)((convN + 255) / 256), 256, 0, stream>>>(W_ih0, W_hh0, W_ihL, W_hhL, Wb);

  // 2) parallel precompute of layer-0 input gates for every timestep.
  k_gi0<<<T_LEN, 256, 0, stream>>>(tokens, emb, Wb /*ih0*/, b_ih0, gi0);

  // 3) persistent bidirectional scan (2 directions * NBLK cooperating blocks).
  k_scan<<<2 * NBLK, 256, 0, stream>>>(tokens, gi0,
      Wb + CH /*hh0*/, Wb + 2 * CH /*ih1*/, Wb + 3 * CH /*hh1*/,
      b_hh0, b_ihL, b_hhL,
      h0f_all, h1f_all, h0b_all, h1b_all,
      gh0a, gi1a, gh1a,
      out, outB, bars);

  // 4) combine directions + final states.
  const size_t combN = (size_t)T_LEN * BH;
  k_combine<<<(unsigned)((combN + 255) / 256), 256, 0, stream>>>(out, outB, h0f_all, h1f_all);
}
